// MambaDecoder_19155554140861
// MI455X (gfx1250) — compile-verified, matrix operand order checked by reading, not run on hardware
//
#include <hip/hip_runtime.h>
#include <math.h>

// ---------------- types ----------------
typedef __bf16 bf16;
typedef __attribute__((ext_vector_type(16))) __bf16 v16bf;
typedef __attribute__((ext_vector_type(8)))  __bf16 v8bf;
typedef __attribute__((ext_vector_type(8)))  float  v8f;

union FragU { v16bf v; v8bf h[2]; };

// ---------------- problem constants ----------------
#define Bc   2
#define Sc   2048
#define Dc   2048
#define Hc   16
#define NOPE 128
#define ROPE 64
#define VD   128
#define KVR  512
#define NE   12
#define TOPK 4
#define EI   1024
#define SEI  2048
#define Tc   (Bc*Sc)         // 4096 tokens
#define QW   (Hc*(NOPE+ROPE))// 3072
#define KVW  (KVR+ROPE)      // 576
#define KVBW (Hc*(NOPE+VD))  // 4096

// ---------------- CDNA5 async global->LDS path ----------------
#if __has_builtin(__builtin_amdgcn_global_load_async_to_lds_b128)
#define HAVE_ASYNC_LDS 1
#else
#define HAVE_ASYNC_LDS 0
#endif

#if HAVE_ASYNC_LDS
typedef int v4i_vs __attribute__((vector_size(16)));
typedef __attribute__((address_space(1))) v4i_vs* as1_v4i_p;
typedef __attribute__((address_space(3))) v4i_vs* as3_v4i_p;

__device__ __forceinline__ void async_copy16(const bf16* g, bf16* l) {
  // GLOBAL_LOAD_ASYNC_TO_LDS_B128: 16 bytes per lane, tracked by ASYNCcnt,
  // no VGPR destination. Builtin signature (from compiler diagnostic):
  //   (int4 AS1* src, int4 AS3* lds_dst, imm offset, imm cpol)
  // Build pointers via uintptr_t: global VA == AS1 address; low 32 bits of a
  // generic LDS address are the LDS offset (AS3 pointers are 32-bit).
  __builtin_amdgcn_global_load_async_to_lds_b128(
      (as1_v4i_p)(uintptr_t)g,
      (as3_v4i_p)(unsigned)(uintptr_t)l,
      0, 0);
}
#endif

__device__ __forceinline__ void wait_async_lds() {
#if HAVE_ASYNC_LDS
#if __has_builtin(__builtin_amdgcn_s_wait_asynccnt)
  __builtin_amdgcn_s_wait_asynccnt(0);
#else
  asm volatile("s_wait_asynccnt 0" ::: "memory");
#endif
#endif
}

// ---------------- GEMM: C[M,N] = alpha * A[M,K] x B[N,K]^T (+causal mask) ----
// A,B bf16 row-major with strides lda/ldb (elements). C fp32, stride ldc.
// 256 threads = 8 wave32s; block tile 128x128, k-tile 32; each wave 64x32.
// Double-buffered LDS with async fills (ping-pong).
#define TM 128
#define TN 128
#define TK 32

__global__ __launch_bounds__(256) void gemm_bf16_wmma(
    const bf16* __restrict__ A, int lda,
    const bf16* __restrict__ Bm, int ldb,
    float* __restrict__ C, int ldc,
    int M, int N, int K, float alpha, int causal)
{
  __shared__ __attribute__((aligned(16))) bf16 As[2][TM][TK];  // 2 x 8KB
  __shared__ __attribute__((aligned(16))) bf16 Bs[2][TN][TK];  // 2 x 8KB

  const int tid  = threadIdx.x;
  const int lane = tid & 31;
  const int wave = tid >> 5;      // 0..7
  const int wm   = wave & 1;      // 2 waves along M
  const int wn   = wave >> 1;     // 4 waves along N
  const int half = lane >> 4;     // K-half select
  const int l16  = lane & 15;

  const int mBlock = blockIdx.y * TM;
  const int nBlock = blockIdx.x * TN;

  v8f acc[4][2];
  for (int i = 0; i < 4; ++i)
    for (int j = 0; j < 2; ++j)
      for (int v = 0; v < 8; ++v) acc[i][j][v] = 0.0f;

  // cooperative tile fill: 16B per thread per iter, 2 iters each for A and B
  auto fill = [&](int buf, int k0) {
    for (int it = 0; it < 2; ++it) {
      int idx = tid + it * 256;         // 0..511
      int row = idx >> 2;               // 0..127
      int chk = idx & 3;                // 4 x 16B chunks per row
      int gm = mBlock + row;
      int gn = nBlock + row;
      const bf16* srcA = A  + (size_t)gm * lda + k0 + chk * 8;
      const bf16* srcB = Bm + (size_t)gn * ldb + k0 + chk * 8;
#if HAVE_ASYNC_LDS
      // OOB rows left unfilled: they only feed accumulator elements whose
      // stores are masked in the epilogue (WMMA (m,n) depends only on A row m
      // and B column n), so garbage never reaches memory.
      if (gm < M) async_copy16(srcA, &As[buf][row][chk * 8]);
      if (gn < N) async_copy16(srcB, &Bs[buf][row][chk * 8]);
#else
      uint4 va = make_uint4(0u, 0u, 0u, 0u);
      if (gm < M) va = *(const uint4*)srcA;
      *(uint4*)&As[buf][row][chk * 8] = va;
      uint4 vb = make_uint4(0u, 0u, 0u, 0u);
      if (gn < N) vb = *(const uint4*)srcB;
      *(uint4*)&Bs[buf][row][chk * 8] = vb;
      __builtin_prefetch(srcA + TK, 0, 1);
      __builtin_prefetch(srcB + TK, 0, 1);
#endif
    }
  };

  const int nk = K / TK;                // K is always a multiple of 32 here
  fill(0, 0);
  wait_async_lds();
  __syncthreads();

  for (int kt = 0; kt < nk; ++kt) {
    const int cur = kt & 1;
    if (kt + 1 < nk) fill(cur ^ 1, (kt + 1) * TK);

    // B fragments (2 n-subtiles), A fragments (4 m-subtiles), 8 WMMAs
    v16bf bfrag[2];
    for (int j = 0; j < 2; ++j) {
      const bf16* rb = &Bs[cur][wn * 32 + j * 16 + l16][0];
      FragU u;
      u.h[0] = *(const v8bf*)(rb + half * 8);
      u.h[1] = *(const v8bf*)(rb + 16 + half * 8);
      bfrag[j] = u.v;
    }
    for (int i = 0; i < 4; ++i) {
      const bf16* ra = &As[cur][wm * 64 + i * 16 + l16][0];
      FragU u;
      u.h[0] = *(const v8bf*)(ra + half * 8);
      u.h[1] = *(const v8bf*)(ra + 16 + half * 8);
      v16bf afrag = u.v;
      for (int j = 0; j < 2; ++j) {
        acc[i][j] = __builtin_amdgcn_wmma_f32_16x16x32_bf16(
            false, afrag, false, bfrag[j], (short)0, acc[i][j], false, false);
      }
    }

    // wait for the next buffer's async fills, then sync all waves before the
    // next iteration overwrites the buffer we just read.
    wait_async_lds();
    __syncthreads();
  }

  // epilogue: C layout -> VGPR v: lanes 0-15 => M=v, N=lane; lanes 16-31 => M=v+8
  for (int i = 0; i < 4; ++i) {
    for (int j = 0; j < 2; ++j) {
      int nCol = nBlock + wn * 32 + j * 16 + l16;
      if (nCol >= N) continue;
      int mBase = mBlock + wm * 64 + i * 16 + half * 8;
      for (int v = 0; v < 8; ++v) {
        int m = mBase + v;
        if (m < M) {
          float val = acc[i][j][v] * alpha;
          if (causal && nCol > m) val += -1.0e9f;
          C[(size_t)m * ldc + nCol] = val;
        }
      }
    }
  }
}

// ---------------- elementwise / helper kernels ----------------
__global__ __launch_bounds__(256) void cast_f32_bf16(const float* __restrict__ s,
                                                     bf16* __restrict__ d, long n) {
  long i = (long)blockIdx.x * 256 + threadIdx.x;
  if (i < n) d[i] = (bf16)s[i];
}

__global__ __launch_bounds__(256) void rmsnorm_to_bf16(const float* __restrict__ in,
                                                       const float* __restrict__ w,
                                                       bf16* __restrict__ out,
                                                       int n, int in_stride) {
  int r = blockIdx.x;
  const float* row = in + (size_t)r * in_stride;
  __shared__ float red[256];
  __shared__ float sc_s;
  float s = 0.0f;
  for (int i = threadIdx.x; i < n; i += 256) { float v = row[i]; s += v * v; }
  red[threadIdx.x] = s; __syncthreads();
  for (int off = 128; off > 0; off >>= 1) {
    if (threadIdx.x < off) red[threadIdx.x] += red[threadIdx.x + off];
    __syncthreads();
  }
  if (threadIdx.x == 0) sc_s = rsqrtf(red[0] / (float)n + 1e-6f);
  __syncthreads();
  float sc = sc_s;
  for (int i = threadIdx.x; i < n; i += 256)
    out[(size_t)r * n + i] = (bf16)(row[i] * sc * w[i]);
}

// RoPE in place on q [T,3072] (cols h*192+128..191) and kv [T,576] (cols 512..575)
__global__ void rope_kernel(float* __restrict__ q, float* __restrict__ kv,
                            const int* __restrict__ start_pos) {
  int t = blockIdx.x;
  int seg = blockIdx.y;               // 0..15 -> q heads, 16 -> k_pe
  int j = threadIdx.x;                // pair index 0..31
  float* p = (seg < 16) ? (q + (size_t)t * QW + seg * (NOPE + ROPE) + NOPE)
                        : (kv + (size_t)t * KVW + KVR);
  float pos = (float)(t % Sc) + (float)start_pos[0];
  float inv = powf(10000.0f, -((float)(2 * j) / (float)ROPE));
  float ang = pos * inv;
  float c = cosf(ang), sn = sinf(ang);
  float t1 = p[2 * j], t2 = p[2 * j + 1];
  p[2 * j]     = t1 * c - t2 * sn;
  p[2 * j + 1] = t1 * sn + t2 * c;
}

// k_bf[t,h,0:128]=kvb[t,h*256+0:128]; k_bf[t,h,128:192]=kv[t,512:576]
__global__ __launch_bounds__(256) void assemble_k(const float* __restrict__ kvb,
                                                  const float* __restrict__ kv,
                                                  bf16* __restrict__ kb) {
  size_t t = blockIdx.x;
  for (int idx = threadIdx.x; idx < Hc * 192; idx += 256) {
    int h = idx / 192, c = idx % 192;
    float v = (c < NOPE) ? kvb[t * KVBW + h * 256 + c]
                         : kv[t * KVW + KVR + (c - NOPE)];
    kb[t * (Hc * 192) + idx] = (bf16)v;
  }
}

// vT[b,h,dv,s] = kvb[(b*S+s), h*256+128+dv]
__global__ __launch_bounds__(256) void assemble_vT(const float* __restrict__ kvb,
                                                   bf16* __restrict__ vT, long n) {
  long i = (long)blockIdx.x * 256 + threadIdx.x;
  if (i >= n) return;
  long s   = i % Sc;
  long rst = i / Sc;
  long dv  = rst % VD;
  long bh  = rst / VD;
  long b = bh / Hc, h = bh % Hc;
  vT[i] = (bf16)kvb[(b * Sc + s) * KVBW + h * 256 + NOPE + dv];
}

// row softmax over n fp32 values; rewrite row in place as bf16
__global__ __launch_bounds__(256) void softmax_inplace_bf16(float* __restrict__ S_, int n) {
  float* row = S_ + (size_t)blockIdx.x * n;
  bf16* outp = (bf16*)row;
  __shared__ float red[256];
  __shared__ float bcast;
  float le[8];
  int per = n / 256;                 // n == 2048 -> 8
  float mx = -1.0e30f;
  for (int k = 0; k < per; ++k) {
    le[k] = row[threadIdx.x + k * 256];
    mx = fmaxf(mx, le[k]);
  }
  red[threadIdx.x] = mx; __syncthreads();
  for (int off = 128; off > 0; off >>= 1) {
    if (threadIdx.x < off) red[threadIdx.x] = fmaxf(red[threadIdx.x], red[threadIdx.x + off]);
    __syncthreads();
  }
  if (threadIdx.x == 0) bcast = red[0];
  __syncthreads();
  float m = bcast;
  float s = 0.0f;
  for (int k = 0; k < per; ++k) { le[k] = expf(le[k] - m); s += le[k]; }
  __syncthreads();
  red[threadIdx.x] = s; __syncthreads();
  for (int off = 128; off > 0; off >>= 1) {
    if (threadIdx.x < off) red[threadIdx.x] += red[threadIdx.x + off];
    __syncthreads();
  }
  if (threadIdx.x == 0) bcast = red[0];
  __syncthreads();
  float inv = 1.0f / bcast;
  for (int k = 0; k < per; ++k)
    outp[threadIdx.x + k * 256] = (bf16)(le[k] * inv);
}

__global__ __launch_bounds__(256) void add_f32(const float* __restrict__ a,
                                               const float* __restrict__ b,
                                               float* __restrict__ o, long n) {
  long i = (long)blockIdx.x * 256 + threadIdx.x;
  if (i < n) o[i] = a[i] + b[i];
}

__global__ __launch_bounds__(256) void silu_mul_bf16(const float* __restrict__ a,
                                                     const float* __restrict__ b,
                                                     bf16* __restrict__ o, long n) {
  long i = (long)blockIdx.x * 256 + threadIdx.x;
  if (i >= n) return;
  float x = a[i];
  float s = x / (1.0f + expf(-x));
  o[i] = (bf16)(s * b[i]);
}

// sigmoid gate, top-4 (by g+bias), normalized weights scattered to we[T,NE]
__global__ __launch_bounds__(256) void gate_topk(const bf16* __restrict__ h2,
                                                 const float* __restrict__ gw,
                                                 const float* __restrict__ gbias,
                                                 float* __restrict__ we) {
  long t = blockIdx.x;
  __shared__ float red[256];
  __shared__ float g[NE];
  for (int e = 0; e < NE; ++e) {
    float s = 0.0f;
    for (int i = threadIdx.x; i < Dc; i += 256)
      s += (float)h2[t * Dc + i] * gw[e * Dc + i];
    red[threadIdx.x] = s; __syncthreads();
    for (int off = 128; off > 0; off >>= 1) {
      if (threadIdx.x < off) red[threadIdx.x] += red[threadIdx.x + off];
      __syncthreads();
    }
    if (threadIdx.x == 0) g[e] = 1.0f / (1.0f + expf(-red[0]));
    __syncthreads();
  }
  if (threadIdx.x == 0) {
    bool used[NE]; for (int e = 0; e < NE; ++e) used[e] = false;
    int   sel_i[TOPK]; float sel_w[TOPK]; float sum = 0.0f;
    for (int k = 0; k < TOPK; ++k) {
      float best = -1.0e30f; int bi = 0;
      for (int e = 0; e < NE; ++e) {
        float v = g[e] + gbias[e];
        if (!used[e] && v > best) { best = v; bi = e; }
      }
      used[bi] = true; sel_i[k] = bi; sel_w[k] = g[bi]; sum += g[bi];
    }
    for (int e = 0; e < NE; ++e) we[t * NE + e] = 0.0f;
    for (int k = 0; k < TOPK; ++k) we[t * NE + sel_i[k]] = sel_w[k] / sum;
  }
}

__global__ __launch_bounds__(256) void scale_accum(float* __restrict__ out,
                                                   const float* __restrict__ y,
                                                   const float* __restrict__ we,
                                                   long n, int e) {
  long i = (long)blockIdx.x * 256 + threadIdx.x;
  if (i >= n) return;
  long t = i / Dc;
  out[i] += we[t * NE + e] * y[i];
}

// ---------------- host orchestration ----------------
static inline dim3 gemm_grid(int M, int N) {
  return dim3((unsigned)((N + TN - 1) / TN), (unsigned)((M + TM - 1) / TM));
}
static inline long cdiv256(long n) { return (n + 255) / 256; }

extern "C" void kernel_launch(void* const* d_in, const int* in_sizes, int n_in,
                              void* d_out, int out_size, void* d_ws, size_t ws_size,
                              hipStream_t stream) {
  const float* x        = (const float*)d_in[0];
  const int*   startpos = (const int*)  d_in[1];
  const float* norm_w   = (const float*)d_in[2];
  const float* wq       = (const float*)d_in[3];
  const float* wkv_a    = (const float*)d_in[4];
  const float* kv_nw    = (const float*)d_in[5];
  const float* wkv_b    = (const float*)d_in[6];
  const float* wo       = (const float*)d_in[7];
  const float* gate_w   = (const float*)d_in[8];
  const float* gate_b   = (const float*)d_in[9];
  const float* w1       = (const float*)d_in[10];
  const float* w2       = (const float*)d_in[11];
  const float* w3       = (const float*)d_in[12];
  const float* sw1      = (const float*)d_in[13];
  const float* sw2      = (const float*)d_in[14];
  const float* sw3      = (const float*)d_in[15];
  float* outp = (float*)d_out;

  char* base = (char*)d_ws;
  size_t off = 0;
  auto alloc = [&](size_t bytes) -> void* {
    void* p = base + off;
    off = (off + bytes + 255) & ~(size_t)255;
    return p;
  };

  bf16* h_bf      = (bf16*)alloc((size_t)Tc * Dc * 2);
  bf16* wq_bf     = (bf16*)alloc((size_t)QW * Dc * 2);
  bf16* wkva_bf   = (bf16*)alloc((size_t)KVW * Dc * 2);
  bf16* wkvb_bf   = (bf16*)alloc((size_t)KVBW * KVR * 2);
  bf16* wo_bf     = (bf16*)alloc((size_t)Dc * Dc * 2);
  bf16* sw1_bf    = (bf16*)alloc((size_t)SEI * Dc * 2);
  bf16* sw2_bf    = (bf16*)alloc((size_t)Dc * SEI * 2);
  bf16* sw3_bf    = (bf16*)alloc((size_t)SEI * Dc * 2);
  bf16* w1_bf     = (bf16*)alloc((size_t)NE * EI * Dc * 2);
  bf16* w2_bf     = (bf16*)alloc((size_t)NE * Dc * EI * 2);
  bf16* w3_bf     = (bf16*)alloc((size_t)NE * EI * Dc * 2);
  float* q        = (float*)alloc((size_t)Tc * QW * 4);
  float* kvv      = (float*)alloc((size_t)Tc * KVW * 4);
  bf16* kvlat_bf  = (bf16*)alloc((size_t)Tc * KVR * 2);
  float* kvb      = (float*)alloc((size_t)Tc * KVBW * 4);
  bf16* qf_bf     = (bf16*)alloc((size_t)Tc * QW * 2);
  bf16* k_bf      = (bf16*)alloc((size_t)Tc * (Hc * 192) * 2);
  bf16* vT_bf     = (bf16*)alloc((size_t)Bc * Hc * VD * Sc * 2);
  float* scores   = (float*)alloc((size_t)Bc * Hc * Sc * Sc * 4);
  float* attn_out = (float*)alloc((size_t)Tc * (Hc * VD) * 4);
  bf16* attn_bf   = (bf16*)alloc((size_t)Tc * (Hc * VD) * 2);
  float* o_proj   = (float*)alloc((size_t)Tc * Dc * 4);
  float* first    = (float*)alloc((size_t)Tc * Dc * 4);
  bf16* h2_bf     = (bf16*)alloc((size_t)Tc * Dc * 2);
  float* t1       = (float*)alloc((size_t)Tc * SEI * 4);
  float* t3       = (float*)alloc((size_t)Tc * SEI * 4);
  bf16* mid_bf    = (bf16*)alloc((size_t)Tc * SEI * 2);
  float* acc_out  = (float*)alloc((size_t)Tc * Dc * 4);
  float* y_e      = (float*)alloc((size_t)Tc * Dc * 4);
  float* we       = (float*)alloc((size_t)Tc * NE * 4);

  // 1) h = RMSNorm(x) -> bf16
  rmsnorm_to_bf16<<<Tc, 256, 0, stream>>>(x, norm_w, h_bf, Dc, Dc);

  // 2) cast weights to bf16
  cast_f32_bf16<<<cdiv256((long)QW * Dc), 256, 0, stream>>>(wq, wq_bf, (long)QW * Dc);
  cast_f32_bf16<<<cdiv256((long)KVW * Dc), 256, 0, stream>>>(wkv_a, wkva_bf, (long)KVW * Dc);
  cast_f32_bf16<<<cdiv256((long)KVBW * KVR), 256, 0, stream>>>(wkv_b, wkvb_bf, (long)KVBW * KVR);
  cast_f32_bf16<<<cdiv256((long)Dc * Dc), 256, 0, stream>>>(wo, wo_bf, (long)Dc * Dc);
  cast_f32_bf16<<<cdiv256((long)SEI * Dc), 256, 0, stream>>>(sw1, sw1_bf, (long)SEI * Dc);
  cast_f32_bf16<<<cdiv256((long)Dc * SEI), 256, 0, stream>>>(sw2, sw2_bf, (long)Dc * SEI);
  cast_f32_bf16<<<cdiv256((long)SEI * Dc), 256, 0, stream>>>(sw3, sw3_bf, (long)SEI * Dc);
  cast_f32_bf16<<<cdiv256((long)NE * EI * Dc), 256, 0, stream>>>(w1, w1_bf, (long)NE * EI * Dc);
  cast_f32_bf16<<<cdiv256((long)NE * Dc * EI), 256, 0, stream>>>(w2, w2_bf, (long)NE * Dc * EI);
  cast_f32_bf16<<<cdiv256((long)NE * EI * Dc), 256, 0, stream>>>(w3, w3_bf, (long)NE * EI * Dc);

  // 3) q = h @ wq^T ; kv = h @ wkv_a^T
  gemm_bf16_wmma<<<gemm_grid(Tc, QW), 256, 0, stream>>>(h_bf, Dc, wq_bf, Dc, q, QW,
                                                        Tc, QW, Dc, 1.0f, 0);
  gemm_bf16_wmma<<<gemm_grid(Tc, KVW), 256, 0, stream>>>(h_bf, Dc, wkva_bf, Dc, kvv, KVW,
                                                         Tc, KVW, Dc, 1.0f, 0);

  // 4) RoPE on q_pe and k_pe (in place)
  rope_kernel<<<dim3(Tc, Hc + 1), 32, 0, stream>>>(q, kvv, startpos);

  // 5) kv_lat = RMSNorm(kv[:, :512]) ; kvb = kv_lat @ wkv_b^T
  rmsnorm_to_bf16<<<Tc, 256, 0, stream>>>(kvv, kv_nw, kvlat_bf, KVR, KVW);
  gemm_bf16_wmma<<<gemm_grid(Tc, KVBW), 256, 0, stream>>>(kvlat_bf, KVR, wkvb_bf, KVR,
                                                          kvb, KVBW, Tc, KVBW, KVR, 1.0f, 0);

  // 6) assemble qf, k, vT
  cast_f32_bf16<<<cdiv256((long)Tc * QW), 256, 0, stream>>>(q, qf_bf, (long)Tc * QW);
  assemble_k<<<Tc, 256, 0, stream>>>(kvb, kvv, k_bf);
  {
    long n = (long)Bc * Hc * VD * Sc;
    assemble_vT<<<cdiv256(n), 256, 0, stream>>>(kvb, vT_bf, n);
  }

  // 7) attention: scores (scale+causal fused), softmax->bf16 in place, PV
  const float scale = 1.0f / sqrtf((float)(NOPE + ROPE));
  for (int b = 0; b < Bc; ++b) {
    for (int h = 0; h < Hc; ++h) {
      const bf16* Aq = qf_bf + ((size_t)b * Sc) * QW + (size_t)h * 192;
      const bf16* Bk = k_bf + ((size_t)b * Sc) * (Hc * 192) + (size_t)h * 192;
      float* Cs = scores + ((size_t)(b * Hc + h)) * Sc * Sc;
      gemm_bf16_wmma<<<gemm_grid(Sc, Sc), 256, 0, stream>>>(Aq, QW, Bk, Hc * 192, Cs, Sc,
                                                            Sc, Sc, 192, scale, 1);
    }
  }
  softmax_inplace_bf16<<<Bc * Hc * Sc, 256, 0, stream>>>(scores, Sc);
  for (int b = 0; b < Bc; ++b) {
    for (int h = 0; h < Hc; ++h) {
      const bf16* Ap = (const bf16*)(scores + ((size_t)(b * Hc + h)) * Sc * Sc);
      const bf16* Bv = vT_bf + ((size_t)(b * Hc + h)) * VD * Sc;
      float* Co = attn_out + ((size_t)b * Sc) * (Hc * VD) + (size_t)h * VD;
      gemm_bf16_wmma<<<gemm_grid(Sc, VD), 256, 0, stream>>>(Ap, 2 * Sc, Bv, Sc, Co, Hc * VD,
                                                            Sc, VD, Sc, 1.0f, 0);
    }
  }

  // 8) o = attn_out @ wo^T ; first = o + x
  cast_f32_bf16<<<cdiv256((long)Tc * Hc * VD), 256, 0, stream>>>(attn_out, attn_bf,
                                                                 (long)Tc * Hc * VD);
  gemm_bf16_wmma<<<gemm_grid(Tc, Dc), 256, 0, stream>>>(attn_bf, Hc * VD, wo_bf, Hc * VD,
                                                        o_proj, Dc, Tc, Dc, Hc * VD, 1.0f, 0);
  add_f32<<<cdiv256((long)Tc * Dc), 256, 0, stream>>>(o_proj, x, first, (long)Tc * Dc);

  // 9) h2 = RMSNorm(first); gating
  rmsnorm_to_bf16<<<Tc, 256, 0, stream>>>(first, norm_w, h2_bf, Dc, Dc);
  gate_topk<<<Tc, 256, 0, stream>>>(h2_bf, gate_w, gate_b, we);

  // 10) shared expert -> acc_out
  gemm_bf16_wmma<<<gemm_grid(Tc, SEI), 256, 0, stream>>>(h2_bf, Dc, sw1_bf, Dc, t1, SEI,
                                                         Tc, SEI, Dc, 1.0f, 0);
  gemm_bf16_wmma<<<gemm_grid(Tc, SEI), 256, 0, stream>>>(h2_bf, Dc, sw3_bf, Dc, t3, SEI,
                                                         Tc, SEI, Dc, 1.0f, 0);
  silu_mul_bf16<<<cdiv256((long)Tc * SEI), 256, 0, stream>>>(t1, t3, mid_bf, (long)Tc * SEI);
  gemm_bf16_wmma<<<gemm_grid(Tc, Dc), 256, 0, stream>>>(mid_bf, SEI, sw2_bf, SEI, acc_out, Dc,
                                                        Tc, Dc, SEI, 1.0f, 0);

  // 11) routed experts (dense, matching reference) with fused weighted accumulate
  for (int e = 0; e < NE; ++e) {
    const bf16* w1e = w1_bf + (size_t)e * EI * Dc;
    const bf16* w3e = w3_bf + (size_t)e * EI * Dc;
    const bf16* w2e = w2_bf + (size_t)e * Dc * EI;
    gemm_bf16_wmma<<<gemm_grid(Tc, EI), 256, 0, stream>>>(h2_bf, Dc, w1e, Dc, t1, EI,
                                                          Tc, EI, Dc, 1.0f, 0);
    gemm_bf16_wmma<<<gemm_grid(Tc, EI), 256, 0, stream>>>(h2_bf, Dc, w3e, Dc, t3, EI,
                                                          Tc, EI, Dc, 1.0f, 0);
    silu_mul_bf16<<<cdiv256((long)Tc * EI), 256, 0, stream>>>(t1, t3, mid_bf, (long)Tc * EI);
    gemm_bf16_wmma<<<gemm_grid(Tc, Dc), 256, 0, stream>>>(mid_bf, EI, w2e, EI, y_e, Dc,
                                                          Tc, Dc, EI, 1.0f, 0);
    scale_accum<<<cdiv256((long)Tc * Dc), 256, 0, stream>>>(acc_out, y_e, we,
                                                            (long)Tc * Dc, e);
  }

  // 12) out = acc_out + first
  add_f32<<<cdiv256((long)Tc * Dc), 256, 0, stream>>>(acc_out, first, outp, (long)Tc * Dc);
}
